// Gemma3Model_31825707664100
// MI455X (gfx1250) — matrix operand order repaired and checked by reading, not compiled, not run on hardware
//
#include <hip/hip_runtime.h>
#include <hip/hip_bf16.h>
#include <math.h>

// ---------------------------------------------------------------------------
// Gemma3-style forward for MI455X (gfx1250, wave32, WMMA).
// All GEMMs: bf16 WMMA (v_wmma_f32_16x16x32_bf16), f32 accumulate.
// Weights stream from HBM once (~485MB fp32 -> ~21us floor at 23.3TB/s),
// converted to bf16 while staging into LDS.
// Fragment loads are arranged as 2x ds_load_b128 per 16x16x32 operand.
// Attention K/V tiles staged with global_load_async_to_lds_b128 (ASYNCcnt).
// ---------------------------------------------------------------------------

typedef __attribute__((ext_vector_type(16))) __bf16 v16bf;
typedef __attribute__((ext_vector_type(8)))  __bf16 v8bf;
typedef __attribute__((ext_vector_type(4)))  __bf16 v4bf;
typedef __attribute__((ext_vector_type(8)))  float  v8f;
typedef __attribute__((ext_vector_type(4)))  float  v4f;

#define S_LEN  1024
#define HDIM   640
#define NHEAD  4
#define HEADD  256
#define NQH    1024     // NHEAD*HEADD
#define IDIM   2048
#define VOCAB  32768
#define NLAYER 18
#define WIN    512
#define RMS_EPS 1e-6f

__device__ inline v8f zero_v8f() {
  v8f z;
#pragma unroll
  for (int i = 0; i < 8; ++i) z[i] = 0.0f;
  return z;
}

union Frag16 { v16bf v; v8bf h[2]; };

// Build a 16x16x32 operand fragment from two contiguous 16B chunks.
__device__ inline v16bf ld_frag2(const __bf16* p0, const __bf16* p1) {
  Frag16 f;
  f.h[0] = *(const v8bf*)p0;
  f.h[1] = *(const v8bf*)p1;
  return f.v;
}

// --- CDNA5 async global->LDS copy (16B per lane), tracked by ASYNCcnt ------
__device__ inline void async_copy16(void* lds_dst, const void* gsrc) {
  unsigned ldsoff = (unsigned)(unsigned long long)lds_dst; // addr[31:0] = LDS offset
  asm volatile("global_load_async_to_lds_b128 %0, %1, off"
               :: "v"(ldsoff), "v"(gsrc) : "memory");
}
__device__ inline void wait_async0() {
  asm volatile("s_wait_asynccnt 0" ::: "memory");
}

// ---------------------------------------------------------------------------
// Embedding gather + sqrt(H) scale
// ---------------------------------------------------------------------------
__global__ __launch_bounds__(256) void embed_kernel(const int* __restrict__ ids,
                                                    const float* __restrict__ embed,
                                                    float* __restrict__ h) {
  int s = blockIdx.x, tid = threadIdx.x;
  int id = ids[s];
  const float sc = 25.298221281347036f; // sqrt(640)
  for (int c = tid; c < HDIM; c += 256)
    h[(size_t)s * HDIM + c] = embed[(size_t)id * HDIM + c] * sc;
}

// ---------------------------------------------------------------------------
// Gemma RMSNorm: out_bf16 = x * rsqrt(mean(x^2)+eps) * (1+w)
// ---------------------------------------------------------------------------
__global__ __launch_bounds__(256) void rmsnorm_bf16_kernel(const float* __restrict__ x,
                                                           const float* __restrict__ w,
                                                           __bf16* __restrict__ out, int ncol) {
  __shared__ float red[256];
  int s = blockIdx.x, tid = threadIdx.x;
  const float* row = x + (size_t)s * ncol;
  float ss = 0.0f;
  for (int c = tid; c < ncol; c += 256) { float v = row[c]; ss += v * v; }
  red[tid] = ss; __syncthreads();
  for (int o = 128; o > 0; o >>= 1) { if (tid < o) red[tid] += red[tid + o]; __syncthreads(); }
  float rinv = rsqrtf(red[0] / (float)ncol + RMS_EPS);
  for (int c = tid; c < ncol; c += 256)
    out[(size_t)s * ncol + c] = (__bf16)(row[c] * rinv * (1.0f + w[c]));
}

// h += rmsnorm(t, w)
__global__ __launch_bounds__(256) void rms_add_kernel(float* __restrict__ h,
                                                      const float* __restrict__ t,
                                                      const float* __restrict__ w, int ncol) {
  __shared__ float red[256];
  int s = blockIdx.x, tid = threadIdx.x;
  const float* row = t + (size_t)s * ncol;
  float ss = 0.0f;
  for (int c = tid; c < ncol; c += 256) { float v = row[c]; ss += v * v; }
  red[tid] = ss; __syncthreads();
  for (int o = 128; o > 0; o >>= 1) { if (tid < o) red[tid] += red[tid + o]; __syncthreads(); }
  float rinv = rsqrtf(red[0] / (float)ncol + RMS_EPS);
  for (int c = tid; c < ncol; c += 256)
    h[(size_t)s * ncol + c] += row[c] * rinv * (1.0f + w[c]);
}

// ---------------------------------------------------------------------------
// Per-head QK RMSNorm + RoPE (theta from device-side is_full flag).
// blockIdx.y: 0..NHEAD-1 -> q heads, NHEAD -> k head. 128 threads, HD=256.
// ---------------------------------------------------------------------------
__global__ __launch_bounds__(128) void qk_rope_kernel(const float* __restrict__ qf,
                                                      const float* __restrict__ kf,
                                                      const float* __restrict__ qn,
                                                      const float* __restrict__ kn,
                                                      __bf16* __restrict__ qb,
                                                      __bf16* __restrict__ kb,
                                                      const unsigned char* __restrict__ is_full,
                                                      int layer) {
  __shared__ float red[128];
  int s = blockIdx.x, hidx = blockIdx.y, tid = threadIdx.x;
  bool isq = hidx < NHEAD;
  const float* src = isq ? (qf + (size_t)s * NQH + (size_t)hidx * HEADD)
                         : (kf + (size_t)s * HEADD);
  const float* wt = isq ? qn : kn;

  float x0 = src[tid];
  float x1 = src[tid + 128];
  red[tid] = x0 * x0 + x1 * x1;
  __syncthreads();
  for (int o = 64; o > 0; o >>= 1) { if (tid < o) red[tid] += red[tid + o]; __syncthreads(); }
  float rinv = rsqrtf(red[0] / (float)HEADD + RMS_EPS);

  float n0 = x0 * rinv * (1.0f + wt[tid]);
  float n1 = x1 * rinv * (1.0f + wt[tid + 128]);

  bool full = is_full[layer] != 0;
  float theta = full ? 1000000.0f : 10000.0f;
  float inv = __expf(-__logf(theta) * ((2.0f * (float)tid) / (float)HEADD));
  float ang = (float)s * inv;
  float c = __cosf(ang), sn = __sinf(ang);
  float o0 = n0 * c - n1 * sn;
  float o1 = n1 * c + n0 * sn;

  __bf16* dst = isq ? (qb + ((size_t)hidx * S_LEN + s) * HEADD)
                    : (kb + (size_t)s * HEADD);
  dst[tid]       = (__bf16)o0;
  dst[tid + 128] = (__bf16)o1;
}

// V: fp32 [S,HD] -> bf16 pre-transposed per 32-key chunk: vbt[S/32][HD][32]
__global__ __launch_bounds__(256) void cvt_v_transpose_kernel(const float* __restrict__ vf,
                                                              __bf16* __restrict__ vbt) {
  int i = blockIdx.x * 256 + threadIdx.x;
  if (i >= S_LEN * HEADD) return;
  int s = i / HEADD, d = i % HEADD;
  int chunk = s >> 5, kin = s & 31;
  vbt[((size_t)chunk * HEADD + d) * 32 + kin] = (__bf16)vf[i];
}

// GeGLU: out = gelu_tanh(g) * u, as bf16
__global__ __launch_bounds__(256) void geglu_kernel(const float* __restrict__ g,
                                                    const float* __restrict__ u,
                                                    __bf16* __restrict__ out, int n) {
  int i = blockIdx.x * 256 + threadIdx.x;
  if (i >= n) return;
  float x = g[i];
  float t = tanhf(0.7978845608028654f * (x + 0.044715f * x * x * x));
  out[i] = (__bf16)(0.5f * x * (1.0f + t) * u[i]);
}

// ---------------------------------------------------------------------------
// bf16 WMMA GEMM: C[M,N] f32 = A[M,K] bf16 @ B (B fp32: [K,N], or [N,K] if BT)
// Block: 256 threads (8 wave32). Tile: 128(M) x 64(N), K-step 32.
// LDS: A K-contiguous, B stored transposed [n][k] -> all fragments are
// pairs of ds_load_b128.
// ---------------------------------------------------------------------------
template <bool BT>
__global__ __launch_bounds__(256) void gemm_bf16_kernel(const __bf16* __restrict__ A,
                                                        const float* __restrict__ Bw,
                                                        float* __restrict__ C,
                                                        int M, int N, int K) {
  __shared__ __bf16 As[128][40];   // [m][k], row 80B (16B-aligned chunks)
  __shared__ __bf16 Bst[64][40];   // [n][k] transposed
  const int tid = threadIdx.x;
  const int lane = tid & 31;
  const int w = tid >> 5;
  const int wm = w >> 1, wn = w & 1;
  const int m0 = blockIdx.y * 128;
  const int n0 = blockIdx.x * 64;

  v8f acc[2][2];
  acc[0][0] = zero_v8f(); acc[0][1] = zero_v8f();
  acc[1][0] = zero_v8f(); acc[1][1] = zero_v8f();

  for (int k0 = 0; k0 < K; k0 += 32) {
    __syncthreads();
    // ---- stage A: 128x32 bf16 as 512 x 16B chunks --------------------------
    for (int i = tid; i < 512; i += 256) {
      int r = i >> 2, cq = i & 3;
      *(v8bf*)&As[r][cq * 8] =
          *(const v8bf*)&A[(size_t)(m0 + r) * K + (k0 + cq * 8)];
    }
    // ---- stage B transposed, fp32->bf16 ------------------------------------
    if constexpr (BT) {   // Bw is [N,K]: contiguous along k -> contiguous store
      for (int i = tid; i < 512; i += 256) {
        int n = i >> 3, kg = i & 7;
        v4f f = *(const v4f*)&Bw[(size_t)(n0 + n) * K + (k0 + kg * 4)];
        v4bf b;
#pragma unroll
        for (int j = 0; j < 4; ++j) b[j] = (__bf16)f[j];
        *(v4bf*)&Bst[n][kg * 4] = b;
      }
    } else {              // Bw is [K,N]: read along n, scatter into columns
      for (int i = tid; i < 512; i += 256) {
        int r = i >> 4, ng = i & 15;
        v4f f = *(const v4f*)&Bw[(size_t)(k0 + r) * N + (n0 + ng * 4)];
#pragma unroll
        for (int j = 0; j < 4; ++j) Bst[ng * 4 + j][r] = (__bf16)f[j];
      }
    }
    if (k0 + 32 < K) {  // global_prefetch_b8 of next K slab
      __builtin_prefetch(&A[(size_t)(m0 + (tid >> 1)) * K + (k0 + 32)], 0, 0);
      if constexpr (BT) __builtin_prefetch(&Bw[(size_t)(n0 + (tid & 63)) * K + (k0 + 32)], 0, 0);
      else              __builtin_prefetch(&Bw[(size_t)(k0 + 32 + (tid & 31)) * N + n0], 0, 0);
    }
    __syncthreads();

    const int mloc = lane & 15;
    const int kb8 = (lane >> 4) << 3;   // A: K base 0/8 (+16 for 2nd chunk)

    v16bf a[2], b[2];
#pragma unroll
    for (int mt = 0; mt < 2; ++mt) {
      const __bf16* pr = &As[wm * 32 + mt * 16 + mloc][0];
      a[mt] = ld_frag2(pr + kb8, pr + 16 + kb8);
    }
#pragma unroll
    for (int nt = 0; nt < 2; ++nt) {
      const __bf16* pc = &Bst[wn * 32 + nt * 16 + mloc][0];
      b[nt] = ld_frag2(pc + kb8, pc + 16 + kb8);
    }
#pragma unroll
    for (int mt = 0; mt < 2; ++mt)
#pragma unroll
      for (int nt = 0; nt < 2; ++nt)
        acc[mt][nt] = __builtin_amdgcn_wmma_f32_16x16x32_bf16(
            false, a[mt], false, b[nt], (short)0, acc[mt][nt], false, false);
  }

  // C/D layout: lane n = lane&15; vgpr v -> row m = v + 8*(lane>=16)
  int hi = (lane >> 4) << 3;
  int cl = lane & 15;
#pragma unroll
  for (int mt = 0; mt < 2; ++mt) {
#pragma unroll
    for (int nt = 0; nt < 2; ++nt) {
      int cn = n0 + wn * 32 + nt * 16 + cl;
      int rb = m0 + wm * 32 + mt * 16 + hi;
#pragma unroll
      for (int v = 0; v < 8; ++v)
        C[(size_t)(rb + v) * N + cn] = acc[mt][nt][v];
    }
  }
}

// ---------------------------------------------------------------------------
// Flash attention: grid (S/64, NHEAD), 128 threads (4 waves x 16 queries).
// K chunk [32][HD] and pre-transposed V chunk [HD][32] staged in LDS via
// global_load_async_to_lds_b128; QK^T and P.V via WMMA bf16; online softmax.
// ---------------------------------------------------------------------------
__global__ __launch_bounds__(128) void attn_kernel(const __bf16* __restrict__ qb,
                                                   const __bf16* __restrict__ kb,
                                                   const __bf16* __restrict__ vbt,
                                                   __bf16* __restrict__ ob,
                                                   const unsigned char* __restrict__ is_full,
                                                   int layer) {
  __shared__ __bf16 Kc[32][HEADD + 8];   // [key][d], row 528B
  __shared__ __bf16 Vt[HEADD][40];       // [d][key], row 80B
  __shared__ float  Sc[4][16][33];
  __shared__ __bf16 Pc[4][16][40];       // probs, row 80B
  __shared__ float  Corr[4][16];
  __shared__ float  Linv[4][16];

  const int tid = threadIdx.x;
  const int lane = tid & 31;
  const int wid = tid >> 5;
  const int head = blockIdx.y;
  const int qblk = blockIdx.x;
  const int q0 = qblk * 64 + wid * 16;
  const bool full = is_full[layer] != 0;
  const float scale = 0.0625f;  // 256^-0.5

  const __bf16* qh = qb + (size_t)head * S_LEN * HEADD;

  // Preload Q fragments (global b128 pairs): 8 K-chunks of 32 along HEADD.
  v16bf aq[8];
  {
    const __bf16* qrow = qh + (size_t)(q0 + (lane & 15)) * HEADD;
    int kb8 = (lane >> 4) << 3;
#pragma unroll
    for (int h = 0; h < 8; ++h)
      aq[h] = ld_frag2(qrow + h * 32 + kb8, qrow + h * 32 + 16 + kb8);
  }

  v8f acc[16];
#pragma unroll
  for (int t = 0; t < 16; ++t) acc[t] = zero_v8f();

  float m_r = -1e30f, l_r = 0.0f;  // per-row running max/sum (lanes 0..15)

  int kstart = 0;
  if (!full) {
    int km = qblk * 64 - (WIN - 1);
    if (km > 0) kstart = (km >> 5) << 5;
  }
  const int kend = qblk * 64 + 64;

  for (int kc = kstart; kc < kend; kc += 32) {
    __syncthreads();
    // ---- async stage K chunk [32][256] and transposed V chunk [256][32] ----
    {
      const __bf16* ksrc = kb + (size_t)kc * HEADD;
      const __bf16* vsrc = vbt + (size_t)(kc >> 5) * HEADD * 32;
      for (int i = tid; i < 1024; i += 128) {
        int r = i >> 5, q = i & 31;            // K: row r, 8-elem group q
        async_copy16(&Kc[r][q * 8], ksrc + r * HEADD + q * 8);
      }
      for (int i = tid; i < 1024; i += 128) {
        int d = i >> 2, q = i & 3;             // V^T: row d, group q
        async_copy16(&Vt[d][q * 8], vsrc + d * 32 + q * 8);
      }
      wait_async0();
    }
    __syncthreads();

    // ---- scores: Q[16x256] . K^T -> two 16x16 tiles ------------------------
    v8f sc2[2];
    sc2[0] = zero_v8f(); sc2[1] = zero_v8f();
    {
      const int kb8 = (lane >> 4) << 3;
      const int nloc = lane & 15;
#pragma unroll
      for (int t = 0; t < 2; ++t) {
        const __bf16* krow = &Kc[t * 16 + nloc][0];
#pragma unroll
        for (int h = 0; h < 8; ++h) {
          v16bf b = ld_frag2(krow + h * 32 + kb8, krow + h * 32 + 16 + kb8);
          sc2[t] = __builtin_amdgcn_wmma_f32_16x16x32_bf16(
              false, aq[h], false, b, (short)0, sc2[t], false, false);
        }
      }
    }
    // ---- scale + mask, stash in LDS ----------------------------------------
    {
      int hi = (lane >> 4) << 3;
      int cl = lane & 15;
#pragma unroll
      for (int t = 0; t < 2; ++t) {
#pragma unroll
        for (int v = 0; v < 8; ++v) {
          int m = v + hi;
          int qq = q0 + m;
          int kk = kc + t * 16 + cl;
          float x = sc2[t][v] * scale;
          bool ok = (kk <= qq) && (full || (qq - kk) < WIN);
          Sc[wid][m][t * 16 + cl] = ok ? x : -1e30f;
        }
      }
    }
    // ---- online softmax (lanes 0..15: one row each) ------------------------
    if (lane < 16) {
      int r = lane;
      float mx = m_r;
#pragma unroll
      for (int c = 0; c < 32; ++c) mx = fmaxf(mx, Sc[wid][r][c]);
      float corr = __expf(m_r - mx);
      float sum = 0.0f;
#pragma unroll
      for (int c = 0; c < 32; ++c) {
        float s = Sc[wid][r][c];
        float p = (s > -1e29f) ? __expf(s - mx) : 0.0f;
        Pc[wid][r][c] = (__bf16)p;
        sum += p;
      }
      m_r = mx;
      l_r = l_r * corr + sum;
      Corr[wid][r] = corr;
    }
    // ---- rescale accumulators ----------------------------------------------
    {
      int hi = (lane >> 4) << 3;
      float f[8];
#pragma unroll
      for (int v = 0; v < 8; ++v) f[v] = Corr[wid][v + hi];
#pragma unroll
      for (int t = 0; t < 16; ++t)
#pragma unroll
        for (int v = 0; v < 8; ++v) acc[t][v] *= f[v];
    }
    // ---- P[16x32] . V[32x256] -> 16 accumulator tiles ----------------------
    {
      const __bf16* prow = &Pc[wid][lane & 15][0];
      const int kb8 = (lane >> 4) << 3;
      v16bf ap = ld_frag2(prow + kb8, prow + 16 + kb8);

      const int nloc = lane & 15;
#pragma unroll
      for (int t = 0; t < 16; ++t) {
        const __bf16* vrow = &Vt[t * 16 + nloc][0];
        v16bf b = ld_frag2(vrow + kb8, vrow + 16 + kb8);
        acc[t] = __builtin_amdgcn_wmma_f32_16x16x32_bf16(
            false, ap, false, b, (short)0, acc[t], false, false);
      }
    }
  }

  if (lane < 16) Linv[wid][lane] = (l_r > 0.0f) ? (1.0f / l_r) : 0.0f;
  {
    int hi = (lane >> 4) << 3;
    int cl = lane & 15;
#pragma unroll
    for (int t = 0; t < 16; ++t) {
#pragma unroll
      for (int v = 0; v < 8; ++v) {
        int m = v + hi;
        float o = acc[t][v] * Linv[wid][m];
        ob[(size_t)(q0 + m) * NQH + (size_t)head * HEADD + t * 16 + cl] = (__bf16)o;
      }
    }
  }
}

// ---------------------------------------------------------------------------
// Host-side orchestration (graph-capture safe; all launches on `stream`)
// ---------------------------------------------------------------------------
extern "C" void kernel_launch(void* const* d_in, const int* in_sizes, int n_in,
                              void* d_out, int out_size, void* d_ws, size_t ws_size,
                              hipStream_t stream) {
  (void)in_sizes; (void)n_in; (void)out_size; (void)ws_size;

  const int*           ids     = (const int*)d_in[0];
  const unsigned char* is_full = (const unsigned char*)d_in[1];
  const float* embed = (const float*)d_in[2];
  const float* wq    = (const float*)d_in[3];
  const float* wk    = (const float*)d_in[4];
  const float* wv    = (const float*)d_in[5];
  const float* wo    = (const float*)d_in[6];
  const float* qn    = (const float*)d_in[7];
  const float* kn    = (const float*)d_in[8];
  const float* ln1   = (const float*)d_in[9];
  const float* ln2   = (const float*)d_in[10];
  const float* ln3   = (const float*)d_in[11];
  const float* ln4   = (const float*)d_in[12];
  const float* wg    = (const float*)d_in[13];
  const float* wu    = (const float*)d_in[14];
  const float* wd    = (const float*)d_in[15];
  const float* normw = (const float*)d_in[16];
  float* out = (float*)d_out;

  char* p = (char*)d_ws;
  auto alloc = [&](size_t bytes) { char* q = p; p += (bytes + 255) & ~(size_t)255; return q; };
  float*  h    = (float*)alloc((size_t)S_LEN * HDIM * 4);
  float*  gbuf = (float*)alloc((size_t)S_LEN * IDIM * 4);
  float*  ubuf = (float*)alloc((size_t)S_LEN * IDIM * 4);
  float*  qf   = (float*)alloc((size_t)S_LEN * NQH * 4);
  float*  kf   = (float*)alloc((size_t)S_LEN * HEADD * 4);
  float*  vf   = (float*)alloc((size_t)S_LEN * HEADD * 4);
  float*  tmpf = (float*)alloc((size_t)S_LEN * HDIM * 4);
  __bf16* xb   = (__bf16*)alloc((size_t)S_LEN * HDIM * 2);
  __bf16* actb = (__bf16*)alloc((size_t)S_LEN * IDIM * 2);
  __bf16* qbv  = (__bf16*)alloc((size_t)NHEAD * S_LEN * HEADD * 2);
  __bf16* kbv  = (__bf16*)alloc((size_t)S_LEN * HEADD * 2);
  __bf16* vbt  = (__bf16*)alloc((size_t)S_LEN * HEADD * 2);  // per-chunk transposed
  __bf16* obv  = (__bf16*)alloc((size_t)S_LEN * NQH * 2);

  embed_kernel<<<S_LEN, 256, 0, stream>>>(ids, embed, h);

  for (int l = 0; l < NLAYER; ++l) {
    const float* wq_l = wq + (size_t)l * HDIM * NQH;
    const float* wk_l = wk + (size_t)l * HDIM * HEADD;
    const float* wv_l = wv + (size_t)l * HDIM * HEADD;
    const float* wo_l = wo + (size_t)l * NQH * HDIM;
    const float* wg_l = wg + (size_t)l * HDIM * IDIM;
    const float* wu_l = wu + (size_t)l * HDIM * IDIM;
    const float* wd_l = wd + (size_t)l * IDIM * HDIM;

    // ---- attention block ----
    rmsnorm_bf16_kernel<<<S_LEN, 256, 0, stream>>>(h, ln1 + (size_t)l * HDIM, xb, HDIM);
    gemm_bf16_kernel<false><<<dim3(NQH / 64, S_LEN / 128), 256, 0, stream>>>(xb, wq_l, qf, S_LEN, NQH, HDIM);
    gemm_bf16_kernel<false><<<dim3(HEADD / 64, S_LEN / 128), 256, 0, stream>>>(xb, wk_l, kf, S_LEN, HEADD, HDIM);
    gemm_bf16_kernel<false><<<dim3(HEADD / 64, S_LEN / 128), 256, 0, stream>>>(xb, wv_l, vf, S_LEN, HEADD, HDIM);
    qk_rope_kernel<<<dim3(S_LEN, NHEAD + 1), 128, 0, stream>>>(qf, kf, qn + (size_t)l * HEADD,
                                                               kn + (size_t)l * HEADD, qbv, kbv, is_full, l);
    cvt_v_transpose_kernel<<<(S_LEN * HEADD + 255) / 256, 256, 0, stream>>>(vf, vbt);
    attn_kernel<<<dim3(S_LEN / 64, NHEAD), 128, 0, stream>>>(qbv, kbv, vbt, obv, is_full, l);
    gemm_bf16_kernel<false><<<dim3(HDIM / 64, S_LEN / 128), 256, 0, stream>>>(obv, wo_l, tmpf, S_LEN, HDIM, NQH);
    rms_add_kernel<<<S_LEN, 256, 0, stream>>>(h, tmpf, ln2 + (size_t)l * HDIM, HDIM);

    // ---- MLP block (GeGLU) ----
    rmsnorm_bf16_kernel<<<S_LEN, 256, 0, stream>>>(h, ln3 + (size_t)l * HDIM, xb, HDIM);
    gemm_bf16_kernel<false><<<dim3(IDIM / 64, S_LEN / 128), 256, 0, stream>>>(xb, wg_l, gbuf, S_LEN, IDIM, HDIM);
    gemm_bf16_kernel<false><<<dim3(IDIM / 64, S_LEN / 128), 256, 0, stream>>>(xb, wu_l, ubuf, S_LEN, IDIM, HDIM);
    geglu_kernel<<<(S_LEN * IDIM + 255) / 256, 256, 0, stream>>>(gbuf, ubuf, actb, S_LEN * IDIM);
    gemm_bf16_kernel<false><<<dim3(HDIM / 64, S_LEN / 128), 256, 0, stream>>>(actb, wd_l, tmpf, S_LEN, HDIM, IDIM);
    rms_add_kernel<<<S_LEN, 256, 0, stream>>>(h, tmpf, ln4 + (size_t)l * HDIM, HDIM);
  }

  // ---- final norm + tied lm-head (embed is [V,H] -> B transposed) ----
  rmsnorm_bf16_kernel<<<S_LEN, 256, 0, stream>>>(h, normw, xb, HDIM);
  gemm_bf16_kernel<true><<<dim3(VOCAB / 64, S_LEN / 128), 256, 0, stream>>>(xb, embed, out, S_LEN, VOCAB, HDIM);
}
